// VanillaLSTMLanguageModel_37366215475650
// MI455X (gfx1250) — matrix operand order repaired and checked by reading, not compiled, not run on hardware
//
#include <hip/hip_runtime.h>
#include <hip/hip_bf16.h>

typedef __bf16 bf16_t;
typedef __attribute__((ext_vector_type(16))) __bf16 v16bf;
typedef __attribute__((ext_vector_type(8)))  __bf16 v8bf;
typedef __attribute__((ext_vector_type(8)))  float  v8f;

constexpr int kVocab = 32000;
constexpr int kH     = 1024;      // hidden
constexpr int kB     = 8;         // batch
constexpr int kS     = 512;       // seq
constexpr int kM     = kB * kS;   // 4096 token rows
constexpr int kG4H   = 4 * kH;    // 4096 gate rows

// ---------------------------------------------------------------------------
// WMMA fragment helpers (CDNA5 16x16x32 bf16, wave32)
// A (16xK): lane L<16 holds row M=L, K=koff..koff+15, koff=0/16 (contiguous)
// B (Kx16): lane L<16 holds col N=L, K=0..15; lanes 16..31 hold K=16..31 (contiguous)
// C/D f32: VGPR r -> M = r + 8*(lane>=16), N = lane&15
// ---------------------------------------------------------------------------
__device__ __forceinline__ v16bf load_a_frag(const bf16_t* row_ptr, int koff) {
  v8bf lo = *(const v8bf*)(row_ptr + koff);
  v8bf hi = *(const v8bf*)(row_ptr + koff + 8);
  return __builtin_shufflevector(lo, hi, 0,1,2,3,4,5,6,7,8,9,10,11,12,13,14,15);
}
__device__ __forceinline__ v16bf load_b_frag(const bf16_t* col_ptr_kbase) {
  v8bf lo = *(const v8bf*)(col_ptr_kbase);
  v8bf hi = *(const v8bf*)(col_ptr_kbase + 8);
  return __builtin_shufflevector(lo, hi, 0,1,2,3,4,5,6,7,8,9,10,11,12,13,14,15);
}

// ---------------------------------------------------------------------------
// gfx1250 async DMA: per-lane 16B copy, global -> LDS, tracked by ASYNCcnt
// ---------------------------------------------------------------------------
__device__ __forceinline__ void async_copy_b128(unsigned lds_byte_off, const void* gptr) {
  asm volatile("global_load_async_to_lds_b128 %0, %1, off"
               :: "v"(lds_byte_off), "v"((unsigned long long)gptr) : "memory");
}
__device__ __forceinline__ void wait_asynccnt0() {
  asm volatile("s_wait_asynccnt 0x0" ::: "memory");
}
__device__ __forceinline__ unsigned lds_off(const void* p) {
  return (unsigned)(uintptr_t)p;   // low 32 bits of generic shared address = LDS offset
}

// ---------------------------------------------------------------------------
// Device-scope generation barrier (persistent-kernel grid sync)
// ---------------------------------------------------------------------------
__device__ __forceinline__ void grid_barrier(unsigned* cnt, unsigned* gen, unsigned nblocks) {
  __syncthreads();
  if (threadIdx.x == 0) {
    __threadfence();
    unsigned my = __hip_atomic_load(gen, __ATOMIC_RELAXED, __HIP_MEMORY_SCOPE_AGENT);
    unsigned arrived =
        __hip_atomic_fetch_add(cnt, 1u, __ATOMIC_ACQ_REL, __HIP_MEMORY_SCOPE_AGENT);
    if (arrived == nblocks - 1u) {
      __hip_atomic_store(cnt, 0u, __ATOMIC_RELAXED, __HIP_MEMORY_SCOPE_AGENT);
      __hip_atomic_fetch_add(gen, 1u, __ATOMIC_RELEASE, __HIP_MEMORY_SCOPE_AGENT);
    } else {
      while (__hip_atomic_load(gen, __ATOMIC_ACQUIRE, __HIP_MEMORY_SCOPE_AGENT) == my) {
        __builtin_amdgcn_s_sleep(1);
      }
    }
    __threadfence();
  }
  __syncthreads();
}

// ---------------------------------------------------------------------------
// Embedding gather: fp32 table -> bf16 activations (also zeroes barrier words)
// ---------------------------------------------------------------------------
__global__ void embed_gather(const int* __restrict__ tokens, const float* __restrict__ Wemb,
                             bf16_t* __restrict__ X, unsigned* __restrict__ barrier) {
  if (blockIdx.x == 0 && threadIdx.x == 0) { barrier[0] = 0u; barrier[1] = 0u; }
  const int row = blockIdx.x;                 // row = b*kS + s
  const int tok = tokens[row];
  const float4* src = (const float4*)(Wemb + (size_t)tok * kH);
  bf16_t* dst = X + (size_t)row * kH;
  for (int i = threadIdx.x; i < kH / 4; i += blockDim.x) {
    float4 f = src[i];
    bf16_t* d = dst + i * 4;
    d[0] = (bf16_t)f.x; d[1] = (bf16_t)f.y; d[2] = (bf16_t)f.z; d[3] = (bf16_t)f.w;
  }
}

// ---------------------------------------------------------------------------
// C[M,N] = A[M,K](bf16) @ W[N,K](f32)^T + bias0[N] + bias1[N]
// 128x128 block tile, double-buffered LDS.
//  - A tile: async DMA global->LDS (ASYNCcnt), bf16 raw
//  - W tile: global f32 loads -> v_cvt_pk_bf16_f32 -> ds_store
// 8 waves, each computes a 32x64 tile = 8 WMMAs per 32-K step; all six
// fragments are loaded into distinct registers before the WMMA chain so the
// matrix pipe runs back-to-back behind a single s_wait_dscnt.
// ---------------------------------------------------------------------------
constexpr int BM = 128, BN = 128, BK = 32, TS = 40;  // TS: padded LDS row stride (bf16)

__global__ __launch_bounds__(256, 2)
void gemm_bf16(const bf16_t* __restrict__ A, const float* __restrict__ W,
               const float* __restrict__ bias0, const float* __restrict__ bias1,
               float* __restrict__ C, int M, int N, int K) {
  __shared__ bf16_t As[2][BM * TS];
  __shared__ bf16_t Ws[2][BN * TS];
  const int tid  = threadIdx.x;
  const int lane = tid & 31;
  const int wid  = tid >> 5;
  const int mbase = blockIdx.y * BM;
  const int nbase = blockIdx.x * BN;
  const int wm = wid & 3;                  // wave M tile (32 rows)
  const int wn = wid >> 2;                 // wave N tile (64 cols)
  const int koff = (lane < 16) ? 0 : 16;   // A-frag K offset
  const int bko  = (lane < 16) ? 0 : 16;   // B-frag K base
  const int colA = lane & 15;

  // W staging map: thread -> (row, 16-col half)
  const int wr_ = tid >> 1;
  const int wc_ = (tid & 1) * 16;

  const int nkb = K / BK;

  auto stage_async_A = [&](int kb, int buf) {
    // 512 16B chunks, 2 per thread: chunk c: row = c>>2, seg = c&3
#pragma unroll
    for (int u = 0; u < 2; ++u) {
      int c   = tid * 2 + u;
      int row = c >> 2;
      int seg = c & 3;
      unsigned l = lds_off(&As[buf][row * TS + seg * 8]);
      const bf16_t* g = A + (size_t)(mbase + row) * K + kb * BK + seg * 8;
      async_copy_b128(l, g);
    }
  };
  auto stage_W = [&](int kb, int buf) {
    const float* gw = W + (size_t)(nbase + wr_) * K + kb * BK + wc_;
    __builtin_prefetch(gw + BK, 0, 1);
    float4 f0 = *(const float4*)(gw);
    float4 f1 = *(const float4*)(gw + 4);
    float4 f2 = *(const float4*)(gw + 8);
    float4 f3 = *(const float4*)(gw + 12);
    bf16_t* d = &Ws[buf][wr_ * TS + wc_];
    d[0]=(bf16_t)f0.x;  d[1]=(bf16_t)f0.y;  d[2]=(bf16_t)f0.z;  d[3]=(bf16_t)f0.w;
    d[4]=(bf16_t)f1.x;  d[5]=(bf16_t)f1.y;  d[6]=(bf16_t)f1.z;  d[7]=(bf16_t)f1.w;
    d[8]=(bf16_t)f2.x;  d[9]=(bf16_t)f2.y;  d[10]=(bf16_t)f2.z; d[11]=(bf16_t)f2.w;
    d[12]=(bf16_t)f3.x; d[13]=(bf16_t)f3.y; d[14]=(bf16_t)f3.z; d[15]=(bf16_t)f3.w;
  };

  v8f acc[2][4];
#pragma unroll
  for (int i = 0; i < 2; ++i)
#pragma unroll
    for (int j = 0; j < 4; ++j) acc[i][j] = (v8f){0.f,0.f,0.f,0.f,0.f,0.f,0.f,0.f};

  int p = 0;
  stage_async_A(0, 0);
  stage_W(0, 0);

  for (int kb = 0; kb < nkb; ++kb) {
    wait_asynccnt0();        // my async copies into As[p] have landed
    __syncthreads();         // => every wave's asyncs + ds_stores for buffer p are done
    if (kb + 1 < nkb) {      // overlap next-stage staging with current compute
      stage_async_A(kb + 1, p ^ 1);
      stage_W(kb + 1, p ^ 1);
    }
    const bf16_t* Ac = As[p];
    const bf16_t* Wc = Ws[p];
    // load all fragments into distinct registers first (one ds clause) ...
    v16bf a0 = load_a_frag(&Ac[(wm * 32 + colA) * TS], koff);
    v16bf a1 = load_a_frag(&Ac[(wm * 32 + 16 + colA) * TS], koff);
    v16bf b0 = load_b_frag(&Wc[(wn * 64 +  0 + colA) * TS + bko]);
    v16bf b1 = load_b_frag(&Wc[(wn * 64 + 16 + colA) * TS + bko]);
    v16bf b2 = load_b_frag(&Wc[(wn * 64 + 32 + colA) * TS + bko]);
    v16bf b3 = load_b_frag(&Wc[(wn * 64 + 48 + colA) * TS + bko]);
    // ... then run the 8-WMMA chain back-to-back on the matrix pipe
    acc[0][0] = __builtin_amdgcn_wmma_f32_16x16x32_bf16(false, a0, false, b0, (short)0, acc[0][0], false, false);
    acc[1][0] = __builtin_amdgcn_wmma_f32_16x16x32_bf16(false, a1, false, b0, (short)0, acc[1][0], false, false);
    acc[0][1] = __builtin_amdgcn_wmma_f32_16x16x32_bf16(false, a0, false, b1, (short)0, acc[0][1], false, false);
    acc[1][1] = __builtin_amdgcn_wmma_f32_16x16x32_bf16(false, a1, false, b1, (short)0, acc[1][1], false, false);
    acc[0][2] = __builtin_amdgcn_wmma_f32_16x16x32_bf16(false, a0, false, b2, (short)0, acc[0][2], false, false);
    acc[1][2] = __builtin_amdgcn_wmma_f32_16x16x32_bf16(false, a1, false, b2, (short)0, acc[1][2], false, false);
    acc[0][3] = __builtin_amdgcn_wmma_f32_16x16x32_bf16(false, a0, false, b3, (short)0, acc[0][3], false, false);
    acc[1][3] = __builtin_amdgcn_wmma_f32_16x16x32_bf16(false, a1, false, b3, (short)0, acc[1][3], false, false);
    p ^= 1;
  }

  const int mo = (lane >> 4) * 8;
#pragma unroll
  for (int j = 0; j < 4; ++j) {
    int gcol = nbase + wn * 64 + j * 16 + colA;
    float bb = (bias0 ? bias0[gcol] : 0.f) + (bias1 ? bias1[gcol] : 0.f);
#pragma unroll
    for (int i = 0; i < 2; ++i) {
#pragma unroll
      for (int rr = 0; rr < 8; ++rr) {
        int grow = mbase + wm * 32 + i * 16 + rr + mo;
        C[(size_t)grow * N + gcol] = acc[i][j][rr] + bb;
      }
    }
  }
}

// ---------------------------------------------------------------------------
// Persistent LSTM layer. 64 blocks x 128 threads (4 waves = 4 gates).
// Wave g keeps its 16 W_hh rows as resident B fragments (256 VGPRs) for all
// 512 steps; each step = stage h to LDS + 32 chained WMMAs + cell update +
// one device-scope barrier (ping-pong h buffers).
// ---------------------------------------------------------------------------
constexpr int HLS = kH + 8;  // padded LDS stride for h tile (bf16 elems)

__global__ __launch_bounds__(128, 1)
void lstm_layer(const float* __restrict__ G, const float* __restrict__ Whh,
                bf16_t* __restrict__ Hseq, bf16_t* __restrict__ hbuf,
                unsigned* __restrict__ barrier) {
  __shared__ bf16_t hlds[16 * HLS];
  __shared__ float  glds[4][16][8];   // [gate][unit][batch]
  const int tid  = threadIdx.x;
  const int lane = tid & 31;
  const int gid  = tid >> 5;                 // gate index 0..3 (i,f,g,o)
  const int ub   = blockIdx.x * 16;          // unit base
  const int koff = (lane < 16) ? 0 : 16;
  const int bko  = (lane < 16) ? 0 : 16;
  const int colA = lane & 15;
  const int nrow = gid * kH + ub + colA;     // gate-row index (W_hh row / G column)

  // --- preload W_hh B-fragments into registers (f32 -> bf16, one-time) ---
  v16bf bfrag[32];
  {
    const float* wr = Whh + (size_t)nrow * kH;
#pragma unroll
    for (int kb = 0; kb < 32; ++kb) {
      const float* p = wr + kb * 32 + bko;
      float4 f0 = *(const float4*)(p);
      float4 f1 = *(const float4*)(p + 4);
      float4 f2 = *(const float4*)(p + 8);
      float4 f3 = *(const float4*)(p + 12);
      v16bf b;
      b[0]=(bf16_t)f0.x;  b[1]=(bf16_t)f0.y;  b[2]=(bf16_t)f0.z;  b[3]=(bf16_t)f0.w;
      b[4]=(bf16_t)f1.x;  b[5]=(bf16_t)f1.y;  b[6]=(bf16_t)f1.z;  b[7]=(bf16_t)f1.w;
      b[8]=(bf16_t)f2.x;  b[9]=(bf16_t)f2.y;  b[10]=(bf16_t)f2.z; b[11]=(bf16_t)f2.w;
      b[12]=(bf16_t)f3.x; b[13]=(bf16_t)f3.y; b[14]=(bf16_t)f3.z; b[15]=(bf16_t)f3.w;
      bfrag[kb] = b;
    }
  }

  // --- zero our columns of both ping-pong h buffers (h0 = 0, padded rows stay 0) ---
  for (int i = tid; i < 2 * 16 * 16; i += blockDim.x) {
    int buf = i >> 8, rrow = (i >> 4) & 15, cc = i & 15;
    hbuf[(size_t)buf * 16 * kH + rrow * kH + ub + cc] = (bf16_t)0.f;
  }

  float creg = 0.f;                 // cell state owned per-thread
  const int uu = tid >> 3;          // unit  0..15
  const int bb = tid & 7;           // batch 0..7

  grid_barrier(barrier, barrier + 1, gridDim.x);

  for (int t = 0; t < kS; ++t) {
    const bf16_t* hread  = hbuf + (size_t)(t & 1) * 16 * kH;
    bf16_t*       hwrite = hbuf + (size_t)((t + 1) & 1) * 16 * kH;

    // stage full h_prev [16 x 1024] into LDS (16B vectors)
    for (int i = tid; i < 16 * (kH / 8); i += blockDim.x) {
      int rrow = i / (kH / 8);
      int cc   = (i % (kH / 8)) * 8;
      *(v8bf*)&hlds[rrow * HLS + cc] = *(const v8bf*)&hread[rrow * kH + cc];
    }
    __syncthreads();

    // gates_hh = h_prev @ Whh_slice^T : 32 chained WMMAs, B resident in VGPRs
    v8f acc = {0.f,0.f,0.f,0.f,0.f,0.f,0.f,0.f};
#pragma unroll
    for (int kb = 0; kb < 32; ++kb) {
      v16bf a = load_a_frag(&hlds[colA * HLS + kb * 32], koff);
      acc = __builtin_amdgcn_wmma_f32_16x16x32_bf16(false, a, false, bfrag[kb],
                                                    (short)0, acc, false, false);
    }

    // add precomputed input gates and exchange through LDS
    if (lane < 16) {
#pragma unroll
      for (int rr = 0; rr < 8; ++rr) {  // rr = batch (lanes<16 hold M=0..7)
        float v = acc[rr] + G[((size_t)(rr * kS + t)) * (size_t)kG4H + nrow];
        glds[gid][colA][rr] = v;
      }
    }
    __syncthreads();

    // LSTM cell update: thread -> (batch bb, unit uu)
    {
      float ig = glds[0][uu][bb];
      float fg = glds[1][uu][bb];
      float gg = glds[2][uu][bb];
      float og = glds[3][uu][bb];
      ig = 1.f / (1.f + __expf(-ig));
      fg = 1.f / (1.f + __expf(-fg));
      og = 1.f / (1.f + __expf(-og));
      gg = tanhf(gg);
      creg = fg * creg + ig * gg;
      float h = og * tanhf(creg);
      int j = ub + uu;
      hwrite[bb * kH + j] = (bf16_t)h;
      Hseq[((size_t)(bb * kS + t)) * kH + j] = (bf16_t)h;
    }

    grid_barrier(barrier, barrier + 1, gridDim.x);
  }
}

// ---------------------------------------------------------------------------
// Host-side orchestration
// ---------------------------------------------------------------------------
extern "C" void kernel_launch(void* const* d_in, const int* in_sizes, int n_in,
                              void* d_out, int out_size, void* d_ws, size_t ws_size,
                              hipStream_t stream) {
  const int*   tokens = (const int*)d_in[0];
  const float* W_emb  = (const float*)d_in[1];
  const float* W_ih   = (const float*)d_in[2];   // [2, 4096, 1024]
  const float* W_hh   = (const float*)d_in[3];   // [2, 4096, 1024]
  const float* b_ih   = (const float*)d_in[4];   // [2, 4096]
  const float* b_hh   = (const float*)d_in[5];   // [2, 4096]
  const float* b_out  = (const float*)d_in[6];   // [32000]
  float* out = (float*)d_out;                    // [4096, 32000]

  char* ws = (char*)d_ws;
  size_t off = 0;
  auto alloc = [&](size_t bytes) -> void* {
    void* p = ws + off;
    off = (off + bytes + 255) & ~(size_t)255;
    return p;
  };
  bf16_t*   X    = (bf16_t*)alloc((size_t)kM * kH * 2);      // embedded tokens (bf16)
  float*    G    = (float*)alloc((size_t)kM * kG4H * 4);     // precomputed input gates
  bf16_t*   H0   = (bf16_t*)alloc((size_t)kM * kH * 2);      // layer-0 outputs (bf16)
  bf16_t*   H1   = (bf16_t*)alloc((size_t)kM * kH * 2);      // layer-1 outputs (bf16)
  bf16_t*   hbuf = (bf16_t*)alloc((size_t)2 * 16 * kH * 2);  // ping-pong h state
  unsigned* bar  = (unsigned*)alloc(256);

  const size_t wstride = (size_t)kG4H * kH;  // per-layer weight stride
  const size_t bstride = (size_t)kG4H;       // per-layer bias stride

  // Phase 1: embedding gather (+ barrier init)
  hipLaunchKernelGGL(embed_gather, dim3(kM), dim3(256), 0, stream, tokens, W_emb, X, bar);

  // Layer 0: input gates (parallel GEMM), then recurrence (persistent WMMA kernel)
  hipLaunchKernelGGL(gemm_bf16, dim3(kG4H / BN, kM / BM), dim3(256), 0, stream,
                     X, W_ih, b_ih, b_hh, G, kM, kG4H, kH);
  hipLaunchKernelGGL(lstm_layer, dim3(64), dim3(128), 0, stream,
                     G, W_hh, H0, hbuf, bar);

  // Layer 1
  hipLaunchKernelGGL(gemm_bf16, dim3(kG4H / BN, kM / BM), dim3(256), 0, stream,
                     H0, W_ih + wstride, b_ih + bstride, b_hh + bstride, G, kM, kG4H, kH);
  hipLaunchKernelGGL(lstm_layer, dim3(64), dim3(128), 0, stream,
                     G, W_hh + wstride, H1, hbuf, bar);

  // Phase 3: tied-embedding logits
  hipLaunchKernelGGL(gemm_bf16, dim3(kVocab / BN, kM / BM), dim3(256), 0, stream,
                     H1, W_emb, b_out, nullptr, out, kM, kVocab, kH);
}
